// EncoderLayer_39616778338908
// MI455X (gfx1250) — compile-verified, matrix operand order checked by reading, not run on hardware
//
#include <hip/hip_runtime.h>
#include <stdint.h>

typedef __attribute__((ext_vector_type(16))) __bf16 v16bf;
typedef __attribute__((ext_vector_type(8)))  float  v8f;
typedef unsigned short u16;
typedef unsigned int   u32;

// ---------------------------------------------------------------- async global->LDS

#if defined(__HIP_DEVICE_COMPILE__) && \
    __has_builtin(__builtin_amdgcn_global_load_async_to_lds_b128) && \
    __has_builtin(__builtin_amdgcn_s_wait_asynccnt)
#define ASYNC_LDS 1
typedef int v4i_b __attribute__((vector_size(16)));
typedef __attribute__((address_space(1))) v4i_b gas_v4i;   // global int4
typedef __attribute__((address_space(3))) v4i_b las_v4i;   // LDS int4
#else
#define ASYNC_LDS 0
#endif

__device__ __forceinline__ void cp_async_b128(u16* lds_dst, const u16* gsrc) {
#if ASYNC_LDS
    __builtin_amdgcn_global_load_async_to_lds_b128(
        (gas_v4i*)(uintptr_t)gsrc,
        (las_v4i*)(u32)(uintptr_t)lds_dst, 0, 0);
#else
    *reinterpret_cast<uint4*>(lds_dst) = *reinterpret_cast<const uint4*>(gsrc);
#endif
}

template<int N>
__device__ __forceinline__ void cp_async_wait() {
#if ASYNC_LDS
    __builtin_amdgcn_s_wait_asynccnt(N);
#endif
}

// ---------------------------------------------------------------- helpers

__device__ __forceinline__ u16 f2bf(float f) {
    union { float f; u32 u; } v; v.f = f;
    u32 r = v.u + 0x7FFFu + ((v.u >> 16) & 1u);   // round-to-nearest-even
    return (u16)(r >> 16);
}

__device__ __forceinline__ v8f vzero8() {
    v8f z;
#pragma unroll
    for (int i = 0; i < 8; i++) z[i] = 0.0f;
    return z;
}

// Load one 16x32 bf16 WMMA fragment (A-layout; identical pattern serves B when
// the matrix is stored [N][K] row-major) from row-major u16 [rows][ld].
// Lane L: row = rowBase + (L&15); elements 0..7  = K kBase+8*half .. +7
//                                 elements 8..15 = K kBase+16+8*half ..
__device__ __forceinline__ v16bf frag_ld(const u16* __restrict__ base, int ld,
                                         int rowBase, int kBase, int lane) {
    const u16* p = base + (size_t)(rowBase + (lane & 15)) * ld + kBase + ((lane >> 4) << 3);
    union { uint4 q[2]; v16bf v; } z;
    z.q[0] = *reinterpret_cast<const uint4*>(p);
    z.q[1] = *reinterpret_cast<const uint4*>(p + 16);
    return z.v;
}

__device__ __forceinline__ v8f wmma_bf(v16bf a, v16bf b, v8f c) {
    return __builtin_amdgcn_wmma_f32_16x16x32_bf16(false, a, false, b, (short)0, c, false, false);
}

// ---------------------------------------------------------------- weight prep

// W: [K][N] f32 row-major  ->  Out: [N][K] bf16 row-major (transposed)
__global__ void wconv_kernel(const float* __restrict__ W, u16* __restrict__ Out,
                             int K, int N, long total) {
    long i = (long)blockIdx.x * 256 + threadIdx.x;
    if (i >= total) return;
    int n = (int)(i / K), k = (int)(i % K);
    Out[i] = f2bf(W[(size_t)k * N + n]);
}

__global__ void bias3_kernel(const float* __restrict__ bq, const float* __restrict__ bk,
                             const float* __restrict__ bv, float* __restrict__ o) {
    int i = blockIdx.x * 256 + threadIdx.x;
    if (i < 768) { o[i] = bq[i]; o[768 + i] = bk[i]; o[1536 + i] = bv[i]; }
}

// ---------------------------------------------------------------- LayerNorm (row=768)

__global__ __launch_bounds__(256) void ln_kernel(const float* __restrict__ in,
                                                 const float* __restrict__ g,
                                                 const float* __restrict__ be,
                                                 u16* __restrict__ out) {
    const int row = blockIdx.x, tid = threadIdx.x, lane = tid & 31, wid = tid >> 5;
    const float* x = in + (size_t)row * 768;
    float v0 = x[tid], v1 = x[tid + 256], v2 = x[tid + 512];
    float s = v0 + v1 + v2;
    float q = v0 * v0 + v1 * v1 + v2 * v2;
#pragma unroll
    for (int m = 1; m < 32; m <<= 1) { s += __shfl_xor(s, m, 32); q += __shfl_xor(q, m, 32); }
    __shared__ float rs[8], rq[8];
    if (lane == 0) { rs[wid] = s; rq[wid] = q; }
    __syncthreads();
    s = rs[lane & 7]; q = rq[lane & 7];
#pragma unroll
    for (int m = 1; m < 8; m <<= 1) { s += __shfl_xor(s, m, 32); q += __shfl_xor(q, m, 32); }
    float mean = s * (1.0f / 768.0f);
    float var  = q * (1.0f / 768.0f) - mean * mean;
    float inv  = rsqrtf(var + 1e-6f);
    size_t o0 = (size_t)row * 768;
    out[o0 + tid]       = f2bf((v0 - mean) * inv * g[tid]       + be[tid]);
    out[o0 + tid + 256] = f2bf((v1 - mean) * inv * g[tid + 256] + be[tid + 256]);
    out[o0 + tid + 512] = f2bf((v2 - mean) * inv * g[tid + 512] + be[tid + 512]);
}

// ---------------------------------------------------------------- GEMM
// C[M][N] = A[M][K](bf16) x Bt[N][K](bf16)^T + bias (+relu) (+resid) -> f32 or bf16
// 128x128 tile per block, BK=32, double-buffered async global->LDS staging,
// 8 waves each computing 32x64 via v_wmma_f32_16x16x32_bf16.

template<bool RELU, bool RESID, bool OUTBF>
__global__ __launch_bounds__(256) void gemm_kernel(const u16* __restrict__ A,
                                                   const u16* __restrict__ Bt,
                                                   const float* __restrict__ bias,
                                                   const float* __restrict__ resid,
                                                   void* __restrict__ outv,
                                                   int M, int N, int K) {
    __shared__ __align__(16) u16 As[2][128 * 40];
    __shared__ __align__(16) u16 Bs[2][128 * 40];
    const int tid = threadIdx.x, lane = tid & 31, wid = tid >> 5;
    const int m0 = blockIdx.y * 128, n0 = blockIdx.x * 128;
    const int wr = (wid >> 1) * 32, wc = (wid & 1) * 64;

    v8f acc[2][4];
#pragma unroll
    for (int i = 0; i < 2; i++)
#pragma unroll
        for (int j = 0; j < 4; j++) acc[i][j] = vzero8();

    const int lr = tid >> 1;          // 0..127 (tile row)
    const int lk = (tid & 1) * 16;    // 0 or 16 (K sub-chunk)
    const size_t aRow = (size_t)(m0 + lr) * K;
    const size_t bRow = (size_t)(n0 + lr) * K;
    const int nk = K >> 5;
    const int lo = lr * 40 + lk;

    // stage tile 0 into buffer 0
    cp_async_b128(&As[0][lo],     A + aRow + lk);
    cp_async_b128(&As[0][lo + 8], A + aRow + lk + 8);
    cp_async_b128(&Bs[0][lo],     Bt + bRow + lk);
    cp_async_b128(&Bs[0][lo + 8], Bt + bRow + lk + 8);

    for (int t = 0; t < nk; t++) {
        const int buf = t & 1;
        if (t + 1 < nk) {   // prefetch next tile into other buffer
            const size_t k0 = (size_t)(t + 1) << 5;
            cp_async_b128(&As[buf ^ 1][lo],     A + aRow + k0 + lk);
            cp_async_b128(&As[buf ^ 1][lo + 8], A + aRow + k0 + lk + 8);
            cp_async_b128(&Bs[buf ^ 1][lo],     Bt + bRow + k0 + lk);
            cp_async_b128(&Bs[buf ^ 1][lo + 8], Bt + bRow + k0 + lk + 8);
            cp_async_wait<4>();     // current tile's 4 copies complete (in-order)
        } else {
            cp_async_wait<0>();
        }
        __syncthreads();

        v16bf af0 = frag_ld(As[buf], 40, wr,      0, lane);
        v16bf af1 = frag_ld(As[buf], 40, wr + 16, 0, lane);
        v16bf bf0 = frag_ld(Bs[buf], 40, wc,      0, lane);
        v16bf bf1 = frag_ld(Bs[buf], 40, wc + 16, 0, lane);
        v16bf bf2 = frag_ld(Bs[buf], 40, wc + 32, 0, lane);
        v16bf bf3 = frag_ld(Bs[buf], 40, wc + 48, 0, lane);

        acc[0][0] = wmma_bf(af0, bf0, acc[0][0]);
        acc[0][1] = wmma_bf(af0, bf1, acc[0][1]);
        acc[0][2] = wmma_bf(af0, bf2, acc[0][2]);
        acc[0][3] = wmma_bf(af0, bf3, acc[0][3]);
        acc[1][0] = wmma_bf(af1, bf0, acc[1][0]);
        acc[1][1] = wmma_bf(af1, bf1, acc[1][1]);
        acc[1][2] = wmma_bf(af1, bf2, acc[1][2]);
        acc[1][3] = wmma_bf(af1, bf3, acc[1][3]);
        __syncthreads();
    }

    const int half8 = (lane >> 4) << 3;
    const int nlo = lane & 15;
#pragma unroll
    for (int rt = 0; rt < 2; rt++)
#pragma unroll
        for (int nt = 0; nt < 4; nt++) {
            int col = n0 + wc + nt * 16 + nlo;
            float bv = bias[col];
#pragma unroll
            for (int e = 0; e < 8; e++) {
                int row = m0 + wr + rt * 16 + half8 + e;
                float v = acc[rt][nt][e] + bv;
                if (RELU) v = fmaxf(v, 0.0f);
                if (RESID) v += resid[(size_t)row * N + col];
                if (OUTBF) ((u16*)outv)[(size_t)row * N + col] = f2bf(v);
                else       ((float*)outv)[(size_t)row * N + col] = v;
            }
        }
}

// ---------------------------------------------------------------- attention
// One block per (b,h,qblock). 8 waves x 32 query rows. Flash-style online softmax.
// qkv: [B*S][2304] bf16 (q|k|v). out: [B*S][768] bf16.

__global__ __launch_bounds__(256) void attn_kernel(const u16* __restrict__ qkv,
                                                   const unsigned char* __restrict__ kpm,
                                                   u16* __restrict__ out) {
    __shared__ __align__(16) u16 Qs[256 * 72];
    __shared__ __align__(16) u16 Ks[256 * 72];
    __shared__ __align__(16) u16 Vts[64 * 264];     // V transposed: [d][key]
    __shared__ __align__(16) u16 Ps[8][32 * 40];    // per-wave P tile (32 rows x 32 keys)
    __shared__ unsigned char Mk[256];

    const int tid = threadIdx.x, lane = tid & 31, wid = tid >> 5;
    const int qb = blockIdx.x & 15;
    const int h  = (blockIdx.x >> 4) % 12;
    const int b  = blockIdx.x / (16 * 12);
    const size_t rowbase = (size_t)b * 4096;
    const int half8 = (lane >> 4) << 3;
    const int nlo = lane & 15;

    {   // stage Q block (256 rows x 64 dims) into LDS (async)
        const u16* src = qkv + (rowbase + qb * 256 + tid) * 2304 + h * 64;
        u16* dst = Qs + tid * 72;
#pragma unroll
        for (int i = 0; i < 8; i++) cp_async_b128(dst + i * 8, src + i * 8);
    }
    cp_async_wait<0>();
    __syncthreads();

    v16bf qa[2][2];
#pragma unroll
    for (int rt = 0; rt < 2; rt++)
#pragma unroll
        for (int ks = 0; ks < 2; ks++)
            qa[rt][ks] = frag_ld(Qs, 72, wid * 32 + rt * 16, ks * 32, lane);

    float mst[2][8], lst[2][8];
    v8f oacc[2][4];
#pragma unroll
    for (int rt = 0; rt < 2; rt++) {
#pragma unroll
        for (int e = 0; e < 8; e++) { mst[rt][e] = -1e30f; lst[rt][e] = 0.0f; }
#pragma unroll
        for (int nt = 0; nt < 4; nt++) oacc[rt][nt] = vzero8();
    }

    for (int kb = 0; kb < 3; kb++) {
        const int gb = qb - 1 + kb;            // uniform across block
        if (gb < 0 || gb > 15) continue;       // out-of-range band block == fully masked
        __syncthreads();
        {   // stage K rows (async), V transposed (scatter), padding mask
            const size_t krow = (rowbase + gb * 256 + tid) * 2304;
            const u16* srck = qkv + krow + 768 + h * 64;
            u16* dstk = Ks + tid * 72;
#pragma unroll
            for (int i = 0; i < 8; i++) cp_async_b128(dstk + i * 8, srck + i * 8);
            const u16* srcv = qkv + krow + 1536 + h * 64;
#pragma unroll
            for (int d = 0; d < 64; d++) Vts[d * 264 + tid] = srcv[d];
            Mk[tid] = kpm[rowbase + gb * 256 + tid];
        }
        cp_async_wait<0>();
        __syncthreads();

#pragma unroll 1
        for (int kc = 0; kc < 8; kc++) {       // 32-key chunks
            v16bf kb0a = frag_ld(Ks, 72, kc * 32,      0,  lane);
            v16bf kb0b = frag_ld(Ks, 72, kc * 32,      32, lane);
            v16bf kb1a = frag_ld(Ks, 72, kc * 32 + 16, 0,  lane);
            v16bf kb1b = frag_ld(Ks, 72, kc * 32 + 16, 32, lane);

            v8f s[2][2];
#pragma unroll
            for (int rt = 0; rt < 2; rt++) {
                v8f c0 = vzero8(), c1 = vzero8();
                c0 = wmma_bf(qa[rt][0], kb0a, c0);
                c0 = wmma_bf(qa[rt][1], kb0b, c0);
                c1 = wmma_bf(qa[rt][0], kb1a, c1);
                c1 = wmma_bf(qa[rt][1], kb1b, c1);
                s[rt][0] = c0; s[rt][1] = c1;
            }

            bool kok0 = (Mk[kc * 32 + nlo] == 0);
            bool kok1 = (Mk[kc * 32 + 16 + nlo] == 0);
            int kj0 = kb * 256 + kc * 32 + nlo;        // band index 0..767
            int kj1 = kj0 + 16;

            float tmax[2][8];
#pragma unroll
            for (int rt = 0; rt < 2; rt++)
#pragma unroll
                for (int e = 0; e < 8; e++) {
                    int qi = wid * 32 + rt * 16 + half8 + e;
                    int d0 = kj0 - qi, d1 = kj1 - qi;
                    bool ok0 = (d0 >= 0) && (d0 <= 512) && kok0;
                    bool ok1 = (d1 >= 0) && (d1 <= 512) && kok1;
                    float v0 = ok0 ? s[rt][0][e] * 0.125f : -1e30f;
                    float v1 = ok1 ? s[rt][1][e] * 0.125f : -1e30f;
                    s[rt][0][e] = v0; s[rt][1][e] = v1;
                    tmax[rt][e] = fmaxf(v0, v1);
                }
            // row max across 16 lanes (within half)
#pragma unroll
            for (int rt = 0; rt < 2; rt++)
#pragma unroll
                for (int e = 0; e < 8; e++) {
                    float t = tmax[rt][e];
#pragma unroll
                    for (int mm = 1; mm < 16; mm <<= 1) t = fmaxf(t, __shfl_xor(t, mm, 32));
                    tmax[rt][e] = t;
                }
            // rescale running state + accumulators
#pragma unroll
            for (int rt = 0; rt < 2; rt++)
#pragma unroll
                for (int e = 0; e < 8; e++) {
                    float nm = fmaxf(mst[rt][e], tmax[rt][e]);
                    float f  = __expf(mst[rt][e] - nm);
                    mst[rt][e] = nm;
                    lst[rt][e] *= f;
#pragma unroll
                    for (int nt = 0; nt < 4; nt++) oacc[rt][nt][e] *= f;
                }
            // P = exp(s - m), accumulate row sums, stage P (bf16) for PV wmma
            float psum[2][8];
#pragma unroll
            for (int rt = 0; rt < 2; rt++)
#pragma unroll
                for (int e = 0; e < 8; e++) psum[rt][e] = 0.0f;
#pragma unroll
            for (int rt = 0; rt < 2; rt++)
#pragma unroll
                for (int nt = 0; nt < 2; nt++)
#pragma unroll
                    for (int e = 0; e < 8; e++) {
                        float sv = s[rt][nt][e];
                        float pv = (sv > -1e29f) ? __expf(sv - mst[rt][e]) : 0.0f;
                        psum[rt][e] += pv;
                        Ps[wid][(rt * 16 + half8 + e) * 40 + nt * 16 + nlo] = f2bf(pv);
                    }
#pragma unroll
            for (int rt = 0; rt < 2; rt++)
#pragma unroll
                for (int e = 0; e < 8; e++) {
                    float ps = psum[rt][e];
#pragma unroll
                    for (int mm = 1; mm < 16; mm <<= 1) ps += __shfl_xor(ps, mm, 32);
                    lst[rt][e] += ps;
                }
            // PV accumulate: O += P(32x32) x V(32x64)
            v16bf pa0 = frag_ld(Ps[wid], 40, 0,  0, lane);
            v16bf pa1 = frag_ld(Ps[wid], 40, 16, 0, lane);
#pragma unroll
            for (int nt = 0; nt < 4; nt++) {
                v16bf vb = frag_ld(Vts, 264, nt * 16, kc * 32, lane);
                oacc[0][nt] = wmma_bf(pa0, vb, oacc[0][nt]);
                oacc[1][nt] = wmma_bf(pa1, vb, oacc[1][nt]);
            }
        }
    }

    // normalize and store [B*S][768] bf16
#pragma unroll
    for (int rt = 0; rt < 2; rt++)
#pragma unroll
        for (int nt = 0; nt < 4; nt++)
#pragma unroll
            for (int e = 0; e < 8; e++) {
                int qi = wid * 32 + rt * 16 + half8 + e;
                float denom = fmaxf(lst[rt][e], 1e-30f);
                float v = oacc[rt][nt][e] / denom;
                out[(rowbase + qb * 256 + qi) * 768 + h * 64 + nt * 16 + nlo] = f2bf(v);
            }
}

// ---------------------------------------------------------------- launch

extern "C" void kernel_launch(void* const* d_in, const int* in_sizes, int n_in,
                              void* d_out, int out_size, void* d_ws, size_t ws_size,
                              hipStream_t stream) {
    (void)in_sizes; (void)n_in; (void)out_size; (void)ws_size;

    const float* x    = (const float*)d_in[0];
    const unsigned char* mask = (const unsigned char*)d_in[1];
    const float* Wq = (const float*)d_in[3];
    const float* bq = (const float*)d_in[4];
    const float* Wk = (const float*)d_in[5];
    const float* bk = (const float*)d_in[6];
    const float* Wv = (const float*)d_in[7];
    const float* bv = (const float*)d_in[8];
    const float* Wo = (const float*)d_in[9];
    const float* bo = (const float*)d_in[10];
    const float* ln1g = (const float*)d_in[11];
    const float* ln1b = (const float*)d_in[12];
    const float* W1 = (const float*)d_in[13];
    const float* b1 = (const float*)d_in[14];
    const float* W2 = (const float*)d_in[15];
    const float* b2 = (const float*)d_in[16];
    const float* ln2g = (const float*)d_in[17];
    const float* ln2b = (const float*)d_in[18];

    const int M = 8192;   // B*S
    char* p = (char*)d_ws;
    auto carve = [&](size_t bytes) -> void* {
        void* r = (void*)p;
        p += (bytes + 255) & ~(size_t)255;
        return r;
    };
    u16*   h1    = (u16*)  carve((size_t)M * 768 * 2);
    u16*   qkv   = (u16*)  carve((size_t)M * 2304 * 2);
    u16*   attn  = (u16*)  carve((size_t)M * 768 * 2);
    float* x2    = (float*)carve((size_t)M * 768 * 4);
    u16*   h2    = (u16*)  carve((size_t)M * 768 * 2);
    u16*   ffn1  = (u16*)  carve((size_t)M * 3072 * 2);
    u16*   Wqkvt = (u16*)  carve((size_t)2304 * 768 * 2);
    float* bqkv  = (float*)carve((size_t)2304 * 4);
    u16*   Wot   = (u16*)  carve((size_t)768 * 768 * 2);
    u16*   W1t   = (u16*)  carve((size_t)3072 * 768 * 2);
    u16*   W2t   = (u16*)  carve((size_t)768 * 3072 * 2);

    // weight prep (bf16, transposed to [N][K])
    long t0 = 768L * 768, t1 = 3072L * 768;
    int g0 = (int)((t0 + 255) / 256), g1 = (int)((t1 + 255) / 256);
    wconv_kernel<<<g0, 256, 0, stream>>>(Wq, Wqkvt,            768, 768, t0);
    wconv_kernel<<<g0, 256, 0, stream>>>(Wk, Wqkvt + t0,       768, 768, t0);
    wconv_kernel<<<g0, 256, 0, stream>>>(Wv, Wqkvt + 2 * t0,   768, 768, t0);
    wconv_kernel<<<g0, 256, 0, stream>>>(Wo, Wot,              768, 768, t0);
    wconv_kernel<<<g1, 256, 0, stream>>>(W1, W1t,              768, 3072, t1);
    wconv_kernel<<<g1, 256, 0, stream>>>(W2, W2t,              3072, 768, t1);
    bias3_kernel<<<3, 256, 0, stream>>>(bq, bk, bv, bqkv);

    // sublayer 1
    ln_kernel<<<M, 256, 0, stream>>>(x, ln1g, ln1b, h1);
    gemm_kernel<false, false, true><<<dim3(2304 / 128, M / 128), 256, 0, stream>>>(
        h1, Wqkvt, bqkv, nullptr, qkv, M, 2304, 768);
    attn_kernel<<<2 * 12 * 16, 256, 0, stream>>>(qkv, mask, attn);
    gemm_kernel<false, true, false><<<dim3(768 / 128, M / 128), 256, 0, stream>>>(
        attn, Wot, bo, x, x2, M, 768, 768);

    // sublayer 2
    ln_kernel<<<M, 256, 0, stream>>>(x2, ln2g, ln2b, h2);
    gemm_kernel<true, false, true><<<dim3(3072 / 128, M / 128), 256, 0, stream>>>(
        h2, W1t, b1, nullptr, ffn1, M, 3072, 768);
    gemm_kernel<false, true, false><<<dim3(768 / 128, M / 128), 256, 0, stream>>>(
        ffn1, W2t, b2, x2, (float*)d_out, M, 768, 3072);
}
